// contextual_layer_45621142618130
// MI455X (gfx1250) — matrix operand order repaired and checked by reading, not compiled, not run on hardware
//
#include <hip/hip_runtime.h>
#include <stdint.h>

// ---------- types ----------
typedef __attribute__((ext_vector_type(16))) __bf16 v16bf;
typedef __attribute__((ext_vector_type(8)))  float  v8f;
typedef __attribute__((ext_vector_type(4)))  unsigned int v4u;
typedef __attribute__((ext_vector_type(8)))  int    v8i;
typedef __attribute__((ext_vector_type(4)))  int    v4i;

#define T_LEN 1024
#define BATCH 64
#define DIMD  256
#define UNITS 256
#define N3    768                 // 3*UNITS
#define FRAGS_PER_MAT (8*48)      // kt(8) x nt(48)
#define FRAG_ELEMS 512            // 32 lanes * 16 bf16
#define MAT_ELEMS (FRAGS_PER_MAT*FRAG_ELEMS)   // 196608 bf16 per matrix

__device__ __forceinline__ v8f wmma_bf16(v16bf a, v16bf b, v8f c) {
  return __builtin_amdgcn_wmma_f32_16x16x32_bf16(false, a, false, b, (short)0, c, false, false);
}
__device__ __forceinline__ float sigmoidf_(float x) { return 1.0f / (1.0f + __expf(-x)); }
__device__ __forceinline__ float tanhf_(float x) {
  float e = __expf(-2.0f * x);
  return (1.0f - e) / (1.0f + e);
}

// ---------------------------------------------------------------------------
// Kernel 0: repack the four f32 [256 x 768] weight matrices into bf16 WMMA
// B-fragment layout: [mat][kt][nt][lane][elem], elem i of lane l holds
// W[kt*32 + (l>>4)*8 + (i&7) + (i>>3)*16][nt*16 + (l&15)]  (16-bit B 32x16 fragment layout).
// ---------------------------------------------------------------------------
__global__ __launch_bounds__(256) void repack_weights(
    const float* __restrict__ kf, const float* __restrict__ kb,
    const float* __restrict__ rf, const float* __restrict__ rb,
    __bf16* __restrict__ out)
{
  int idx = blockIdx.x * 256 + threadIdx.x;      // 4*8*48*512 = 786432 total
  int e    = idx & 15;
  int l    = (idx >> 4) & 31;
  int frag = idx >> 9;                           // [0, 1536)
  int nt   = frag % 48;
  int kt   = (frag / 48) & 7;
  int mat  = frag / FRAGS_PER_MAT;
  const float* W = (mat == 0) ? kf : (mat == 1) ? kb : (mat == 2) ? rf : rb;
  int k = kt * 32 + ((l >> 4) * 8) + (e & 7) + ((e >> 3) * 16);
  int n = nt * 16 + (l & 15);
  out[idx] = (__bf16)W[(size_t)k * N3 + n];
}

// ---------------------------------------------------------------------------
// Kernel 1: input projection GEMM (bf16 WMMA, f32 acc).
//   xw_dir[t', b, col] = x[b, t, :] @ kernel_dir + bias_dir[0]
// with t' = t (fwd) or 1023-t (bwd, pre-reversed so the scan runs forward).
// Grid: (M/128, 6).  WG = 8 waves; wave = 16 M-rows x 256 N-cols (16 tiles).
// M index m = t*64 + b.
// ---------------------------------------------------------------------------
__global__ __launch_bounds__(256) void input_gemm(
    const float* __restrict__ x,
    const __bf16* __restrict__ wfrag,      // mats 0 (kernel_f), 1 (kernel_b)
    const float* __restrict__ bias_f, const float* __restrict__ bias_b,
    float* __restrict__ xw_f, float* __restrict__ xw_b)
{
  const int lane = threadIdx.x & 31;
  const int w    = threadIdx.x >> 5;
  const int m0   = blockIdx.x * 128 + w * 16;
  const int n0   = blockIdx.y * 256;             // 0..1535
  const int dir  = (n0 >= N3);
  const int nbase = dir ? (n0 - N3) : n0;        // 0 / 256 / 512
  const __bf16* wf   = wfrag + (size_t)dir * MAT_ELEMS;
  const float* bias0 = dir ? bias_b : bias_f;    // row 0 of bias[2][768]
  float* outp        = dir ? xw_b : xw_f;

  // A-row for this lane (rows m0..m0+15 live in lanes &15)
  const int mrow  = m0 + (lane & 15);
  const int tt    = mrow >> 6;
  const int bb    = mrow & 63;
  const float* xrow = x + ((size_t)bb * T_LEN + tt) * DIMD;
  const int khalf = (lane >> 4) * 8;

  v8f acc[16];
#pragma unroll
  for (int i = 0; i < 16; ++i) acc[i] = (v8f){};

#pragma unroll
  for (int kt = 0; kt < 8; ++kt) {
    // A fragment: a[0..7] = x[m, k0..k0+7], a[8..15] = x[m, k0+16..k0+23]
    const float* ap = xrow + kt * 32 + khalf;
    float4 f0 = *(const float4*)(ap);
    float4 f1 = *(const float4*)(ap + 4);
    float4 f2 = *(const float4*)(ap + 16);
    float4 f3 = *(const float4*)(ap + 20);
    v16bf a;
    a[0]=(__bf16)f0.x; a[1]=(__bf16)f0.y; a[2]=(__bf16)f0.z; a[3]=(__bf16)f0.w;
    a[4]=(__bf16)f1.x; a[5]=(__bf16)f1.y; a[6]=(__bf16)f1.z; a[7]=(__bf16)f1.w;
    a[8]=(__bf16)f2.x; a[9]=(__bf16)f2.y; a[10]=(__bf16)f2.z; a[11]=(__bf16)f2.w;
    a[12]=(__bf16)f3.x; a[13]=(__bf16)f3.y; a[14]=(__bf16)f3.z; a[15]=(__bf16)f3.w;
#pragma unroll
    for (int nt = 0; nt < 16; ++nt) {
      const __bf16* bp = wf + ((size_t)(kt * 48 + (nbase >> 4) + nt)) * FRAG_ELEMS + lane * 16;
      v16bf bfr = *(const v16bf*)bp;
      acc[nt] = wmma_bf16(a, bfr, acc[nt]);
    }
  }

  // bias + scatter to xw[t', b, col]
#pragma unroll
  for (int nt = 0; nt < 16; ++nt) {
    const int col = nbase + nt * 16 + (lane & 15);
    const float bv = bias0[col];
#pragma unroll
    for (int j = 0; j < 8; ++j) {
      const int m = m0 + j + 8 * (lane >> 4);
      const int t = m >> 6, b = m & 63;
      const int tp = dir ? (T_LEN - 1 - t) : t;
      outp[((size_t)tp * BATCH + b) * N3 + col] = acc[nt][j] + bv;
    }
  }
}

// ---------------------------------------------------------------------------
// TDM: async-copy one step's xw tile (16 rows x 768 f32 = 48KB contiguous)
// into LDS.  1-D descriptor: data_size=4B, tile_dim0 = tensor_dim0 = 12288.
// 6-arg builtin form: (v4u g0, v8i g1, v4i g2, v4i g3, v8i g4, i32 cpol).
// ---------------------------------------------------------------------------
__device__ __forceinline__ void tdm_issue_xw(const float* gsrc, uint32_t ldsoff) {
  const uint64_t ga = (uint64_t)gsrc;
  v4u g0 = { 1u,                                   // count=1, user descriptor
             ldsoff,                               // lds_addr (bytes)
             (uint32_t)ga,                         // global_addr[31:0]
             (uint32_t)((ga >> 32) & 0x01FFFFFFu) | (2u << 30) };  // addr[56:32] | type=2
  v8i g1 = { (int)(2u << 16),                      // data_size = 4 bytes
             (int)(12288u << 16),                  // tensor_dim0 lo16 << 16
             (int)(1u << 16),                      // tensor_dim0 hi16=0 | tensor_dim1=1
             (int)(12288u << 16),                  // tensor_dim1 hi=0 | tile_dim0=12288
             (int)1,                               // tile_dim1=1, tile_dim2=0
             (int)12288,                           // tensor_dim0_stride lo32
             0, 0 };
  v4i g2 = {0, 0, 0, 0};
  v4i g3 = {0, 0, 0, 0};
  v8i g4 = {0, 0, 0, 0, 0, 0, 0, 0};
  __builtin_amdgcn_tensor_load_to_lds(g0, g1, g2, g3, g4, 0);
}

// ---------------------------------------------------------------------------
// Kernel 2: persistent GRU scan.  Grid (4 batch-groups, 2 directions),
// 512 threads = 16 waves.  Wave w owns hidden slice u in [16w,16w+16) and
// rec-weight tiles {w, w+16, w+32} (z,r,h) -> 24 register-resident bf16
// B-fragments.  Per step: 24 WMMAs/wave, gates in-registers, h round-trips
// through an 8KB LDS buffer as next step's A operand.  xw tiles are
// TDM-prefetched (double buffered) and waited with s_wait_tensorcnt.
// ---------------------------------------------------------------------------
__global__ __launch_bounds__(512) void gru_scan(
    const __bf16* __restrict__ recfrag,    // mats: [0]=rec_f, [1]=rec_b (repacked)
    const float* __restrict__ bias_f, const float* __restrict__ bias_b,
    const float* __restrict__ xw_f, const float* __restrict__ xw_b,
    float* __restrict__ out)               // [64,1024,512]
{
  __shared__ float  xwbuf[2][16 * N3];     // 2 x 48KB
  __shared__ __bf16 hlds[16 * UNITS];      // 8KB

  const int tid  = threadIdx.x;
  const int lane = tid & 31;
  const int w    = tid >> 5;               // 0..15
  const int g    = blockIdx.x;             // batch group 0..3
  const int dir  = blockIdx.y;             // 0 fwd, 1 bwd
  const int b0   = g * 16;

  const __bf16* rf = recfrag + (size_t)dir * MAT_ELEMS;
  const float*  b1 = (dir ? bias_b : bias_f) + N3;   // bias row 1
  const float*  xwsrc = dir ? xw_b : xw_f;

  // zero h state in LDS
  for (int i = tid; i < 16 * UNITS; i += 512) hlds[i] = (__bf16)0.0f;

  // register-resident rec-weight fragments (z, r, h tiles of this wave)
  v16bf bz[8], br[8], bh[8];
#pragma unroll
  for (int kt = 0; kt < 8; ++kt) {
    bz[kt] = *(const v16bf*)(rf + (size_t)(kt * 48 + w     ) * FRAG_ELEMS + lane * 16);
    br[kt] = *(const v16bf*)(rf + (size_t)(kt * 48 + 16 + w) * FRAG_ELEMS + lane * 16);
    bh[kt] = *(const v16bf*)(rf + (size_t)(kt * 48 + 32 + w) * FRAG_ELEMS + lane * 16);
  }

  const int u = w * 16 + (lane & 15);      // hidden unit of this lane's column
  const float b1z = b1[u], b1r = b1[UNITS + u], b1h = b1[2 * UNITS + u];

  float h[8];
#pragma unroll
  for (int j = 0; j < 8; ++j) h[j] = 0.0f;

  const uint32_t lds0 = (uint32_t)(uintptr_t)&xwbuf[0][0];
  const uint32_t lds1 = (uint32_t)(uintptr_t)&xwbuf[1][0];

  if (w == 0) tdm_issue_xw(xwsrc + (size_t)b0 * N3, lds0);
  __syncthreads();                          // hlds zeros visible

  for (int t = 0; t < T_LEN; ++t) {
    // prefetch next step's xw tile into the other buffer
    if (w == 0 && (t + 1) < T_LEN)
      tdm_issue_xw(xwsrc + ((size_t)(t + 1) * BATCH + b0) * N3, (t & 1) ? lds0 : lds1);

    // rec GEMM: [16 x 256] (h, bf16) x [256 x 48] (this wave's z/r/h tiles)
    v8f cz = (v8f){}, cr = (v8f){}, ch = (v8f){};
#pragma unroll
    for (int kt = 0; kt < 8; ++kt) {
      const __bf16* hp = hlds + (lane & 15) * UNITS + kt * 32 + (lane >> 4) * 8;
      union { v16bf v; uint4 q[2]; } au;
      au.q[0] = *(const uint4*)hp;
      au.q[1] = *(const uint4*)(hp + 16);
      cz = wmma_bf16(au.v, bz[kt], cz);
      cr = wmma_bf16(au.v, br[kt], cr);
      ch = wmma_bf16(au.v, bh[kt], ch);
    }

    // wait for this step's xw tile (next step's TDM may still be in flight)
    if (w == 0) {
      if ((t + 1) < T_LEN) __builtin_amdgcn_s_wait_tensorcnt(1);
      else                 __builtin_amdgcn_s_wait_tensorcnt(0);
    }
    __syncthreads();   // xw tile ready for all; all waves done reading hlds

    const float* xb = xwbuf[t & 1];
    const int tout = dir ? (T_LEN - 1 - t) : t;
#pragma unroll
    for (int j = 0; j < 8; ++j) {
      const int m = j + 8 * (lane >> 4);                 // local batch row
      const float xz = xb[m * N3 + u];
      const float xr = xb[m * N3 + UNITS + u];
      const float xh = xb[m * N3 + 2 * UNITS + u];
      const float z  = sigmoidf_(xz + cz[j] + b1z);
      const float r  = sigmoidf_(xr + cr[j] + b1r);
      const float hh = tanhf_(xh + r * (ch[j] + b1h));
      const float hn = z * h[j] + (1.0f - z) * hh;
      h[j] = hn;
      out[((size_t)(b0 + m) * T_LEN + tout) * 512 + dir * UNITS + u] = hn;
      hlds[m * UNITS + u] = (__bf16)hn;
    }
    __syncthreads();   // h_new visible before next step's GEMM
  }
}

// ---------------------------------------------------------------------------
// launch
// ---------------------------------------------------------------------------
extern "C" void kernel_launch(void* const* d_in, const int* in_sizes, int n_in,
                              void* d_out, int out_size, void* d_ws, size_t ws_size,
                              hipStream_t stream) {
  const float* x    = (const float*)d_in[0];
  const float* kf   = (const float*)d_in[1];
  const float* rf   = (const float*)d_in[2];
  const float* bf   = (const float*)d_in[3];
  const float* kb   = (const float*)d_in[4];
  const float* rb   = (const float*)d_in[5];
  const float* bb   = (const float*)d_in[6];
  float* out = (float*)d_out;

  char* ws = (char*)d_ws;
  __bf16* wfrag = (__bf16*)ws;                                  // 4 * 384KB
  float* xw_f = (float*)(ws + (size_t)4 * MAT_ELEMS * sizeof(__bf16));
  float* xw_b = xw_f + (size_t)T_LEN * BATCH * N3;

  // 0) repack weights into bf16 WMMA fragment layout
  repack_weights<<<(4 * MAT_ELEMS) / 256, 256, 0, stream>>>(kf, kb, rf, rb, wfrag);

  // 1) input projection GEMM (both directions; bwd written time-reversed)
  input_gemm<<<dim3((BATCH * T_LEN) / 128, 6), 256, 0, stream>>>(
      x, wfrag, bf, bb, xw_f, xw_b);

  // 2) persistent recurrent scan: 4 batch groups x 2 directions
  gru_scan<<<dim3(4, 2), 512, 0, stream>>>(
      wfrag + (size_t)2 * MAT_ELEMS, bf, bb, xw_f, xw_b, out);
}
